// ProtoNet_28329604284467
// MI455X (gfx1250) — matrix operand order repaired and checked by reading, not compiled, not run on hardware
//
#include <hip/hip_runtime.h>
#include <hip/hip_bf16.h>
#include <hip/hip_fp16.h>

// ---------------- problem constants (match reference) ----------------
#define NN   100000
#define EE   3200000
#define FIN  128
#define EDIM 12
#define HC   64      // H*C for both layers
#define GG   64

typedef __attribute__((ext_vector_type(16))) _Float16 v16h;
typedef __attribute__((ext_vector_type(8)))  float    v8f;

// ---------------- helpers ----------------
__device__ __forceinline__ unsigned fenc(float f) {
    unsigned u = __float_as_uint(f);
    return (u & 0x80000000u) ? ~u : (u | 0x80000000u);   // order-preserving
}
__device__ __forceinline__ float fdec(unsigned k) {
    unsigned u = (k & 0x80000000u) ? (k ^ 0x80000000u) : ~k;
    return __uint_as_float(u);
}

__global__ void fill_zero(float* __restrict__ p, long n) {
    long t = (long)blockIdx.x * blockDim.x + threadIdx.x;
    if (t < n) p[t] = 0.f;
}

// ---------------- self-loop mean edge-attr ----------------
__global__ void sl_accum(const int* __restrict__ ei, const float* __restrict__ eattr,
                         float* __restrict__ loop_attr, long total) {
    long t = (long)blockIdx.x * blockDim.x + threadIdx.x;
    if (t >= total) return;
    long e = t / EDIM; int k = (int)(t % EDIM);
    int d = ei[(long)EE + e];
    atomicAdd(&loop_attr[(size_t)d * EDIM + k], eattr[e * (long)EDIM + k]);
}
__global__ void deg_accum(const int* __restrict__ ei, float* __restrict__ deg, long E) {
    long t = (long)blockIdx.x * blockDim.x + threadIdx.x;
    if (t >= E) return;
    atomicAdd(&deg[ei[(long)EE + t]], 1.f);
}
__global__ void sl_div(float* __restrict__ loop_attr, const float* __restrict__ deg, long total) {
    long t = (long)blockIdx.x * blockDim.x + threadIdx.x;
    if (t >= total) return;
    loop_attr[t] /= fmaxf(deg[t / EDIM], 1.f);
}

// ---------------- WMMA GEMM: C[n,64] = A[n,K] @ W[K,64] ----------------
// 256 threads = 8 waves per block. W staged once into LDS (f16, col-major).
// Each wave owns a 16-row x 64-col strip: one A fragment feeds 4 WMMAs/k-step.
template <int K>
__global__ void __launch_bounds__(256) gemm_wmma(const float* __restrict__ A,
                                                 const float* __restrict__ Wg,
                                                 float* __restrict__ C, int nrows) {
    __shared__ __align__(32) _Float16 Wt[64 * K];    // Wt[c*K + k] = W[k][c]
    for (int idx = threadIdx.x; idx < K * 64; idx += 256) {
        int k = idx >> 6, c = idx & 63;              // coalesced global read over c
        Wt[c * K + k] = (_Float16)Wg[idx];
    }
    __syncthreads();

    int waveId = threadIdx.x >> 5;
    int lane   = threadIdx.x & 31;
    int tileM  = (blockIdx.x * 8 + waveId) * 16;
    if (tileM >= nrows) return;                      // after barrier: safe

    int row     = tileM + (lane & 15);
    int colBase = lane & 15;
    int hiHalf  = (lane & 16) ? 1 : 0;

    v8f acc[4] = {};
    for (int kk = 0; kk < K; kk += 32) {
        // A fragment 16x32 f16: lane<16 -> K kk+0..15; lane>=16 -> kk+16..31
        v16h a;
        const float* ap = A + (size_t)row * K + kk + (hiHalf ? 16 : 0);
#pragma unroll
        for (int i = 0; i < 8; ++i) {
            a[i]     = (_Float16)ap[i];
            a[i + 8] = (_Float16)ap[i + 8];
        }
        int kb = kk + (hiHalf ? 16 : 0);             // B: lane<16 -> K kk+0..15; else +16..31
#pragma unroll
        for (int tn = 0; tn < 4; ++tn) {
            v16h b = *(const v16h*)&Wt[(size_t)(tn * 16 + colBase) * K + kb]; // 2x ds_load_b128
            acc[tn] = __builtin_amdgcn_wmma_f32_16x16x32_f16(
                false, a, false, b, (short)0, acc[tn], false, false);
        }
    }
    // C layout: VGPR j -> M = tileM + j + 8*hiHalf, N = tn*16 + (lane&15)
    int rbase = tileM + (hiHalf ? 8 : 0);
#pragma unroll
    for (int tn = 0; tn < 4; ++tn)
#pragma unroll
        for (int j = 0; j < 8; ++j)
            C[(size_t)(rbase + j) * HC + tn * 16 + colBase] = acc[tn][j];
}

// ---------------- per-node attention terms ----------------
__global__ void node_alpha(const float* __restrict__ xl, const float* __restrict__ att_s,
                           const float* __restrict__ att_d, float* __restrict__ as,
                           float* __restrict__ ad, long n, int H, int C) {
    long t = (long)blockIdx.x * blockDim.x + threadIdx.x;
    if (t >= n * H) return;
    long node = t / H; int h = (int)(t % H);
    const float* v = xl + (size_t)node * HC + h * C;
    float ss = 0.f, sd = 0.f;
    for (int c = 0; c < C; ++c) {
        ss += v[c] * att_s[h * C + c];
        sd += v[c] * att_d[h * C + c];
    }
    as[t] = ss; ad[t] = sd;
}

// M[k,h] = sum_c We[k, h*C+c] * att_e[h,c]
__global__ void make_M(const float* __restrict__ We, const float* __restrict__ ae,
                       float* __restrict__ M, int H, int C) {
    int t = blockIdx.x * blockDim.x + threadIdx.x;
    if (t >= EDIM * H) return;
    int k = t / H, h = t % H;
    float s = 0.f;
    for (int c = 0; c < C; ++c) s += We[k * HC + h * C + c] * ae[h * C + c];
    M[k * H + h] = s;
}

// ---------------- edge passes ----------------
__device__ __forceinline__ void edec(const int* __restrict__ ei, const float* __restrict__ eattr,
                                     const float* __restrict__ loop_attr, long e,
                                     int& s, int& d, const float*& ea) {
    if (e < EE) {
        s = ei[e]; d = ei[(long)EE + e];
        ea = eattr + e * (long)EDIM;
        if (e + 64 < EE) __builtin_prefetch(eattr + (e + 64) * (long)EDIM, 0, 1);
    } else {
        int v = (int)(e - EE); s = v; d = v;
        ea = loop_attr + (size_t)v * EDIM;
    }
}

__device__ __forceinline__ float edge_alpha(const float* __restrict__ ea,
                                            const float* __restrict__ M,
                                            const float* __restrict__ as,
                                            const float* __restrict__ ad,
                                            int s, int d, int h, int H) {
    float x = 0.f;
#pragma unroll
    for (int k = 0; k < EDIM; ++k) x += ea[k] * M[k * H + h];
    x += as[(size_t)s * H + h] + ad[(size_t)d * H + h];
    return x > 0.f ? x : 0.2f * x;   // leaky_relu(0.2)
}

__global__ void edge_amax_k(const int* __restrict__ ei, const float* __restrict__ eattr,
                            const float* __restrict__ loop_attr, const float* __restrict__ M,
                            const float* __restrict__ as, const float* __restrict__ ad,
                            unsigned* __restrict__ amax, long Etot, int H) {
    long t = (long)blockIdx.x * blockDim.x + threadIdx.x;
    if (t >= Etot * H) return;
    long e = t / H; int h = (int)(t % H);
    int s, d; const float* ea;
    edec(ei, eattr, loop_attr, e, s, d, ea);
    float a = edge_alpha(ea, M, as, ad, s, d, h, H);
    atomicMax(&amax[(size_t)d * H + h], fenc(a));
}

__global__ void edge_den_k(const int* __restrict__ ei, const float* __restrict__ eattr,
                           const float* __restrict__ loop_attr, const float* __restrict__ M,
                           const float* __restrict__ as, const float* __restrict__ ad,
                           const unsigned* __restrict__ amax, float* __restrict__ den,
                           long Etot, int H) {
    long t = (long)blockIdx.x * blockDim.x + threadIdx.x;
    if (t >= Etot * H) return;
    long e = t / H; int h = (int)(t % H);
    int s, d; const float* ea;
    edec(ei, eattr, loop_attr, e, s, d, ea);
    float a = edge_alpha(ea, M, as, ad, s, d, h, H);
    float ex = __expf(a - fdec(amax[(size_t)d * H + h]));
    atomicAdd(&den[(size_t)d * H + h], ex);
}

// one thread per (edge, 8-channel group); H==8 -> group == head, H==1 -> shared head 0
__global__ void edge_agg_k(const int* __restrict__ ei, const float* __restrict__ eattr,
                           const float* __restrict__ loop_attr, const float* __restrict__ M,
                           const float* __restrict__ as, const float* __restrict__ ad,
                           const unsigned* __restrict__ amax, const float* __restrict__ den,
                           const float* __restrict__ xl, float* __restrict__ out,
                           long Etot, int H) {
    long t = (long)blockIdx.x * blockDim.x + threadIdx.x;
    if (t >= (Etot << 3)) return;
    long e = t >> 3; int g = (int)(t & 7);
    int h = (H == 8) ? g : 0;
    int s, d; const float* ea;
    edec(ei, eattr, loop_attr, e, s, d, ea);
    float a = edge_alpha(ea, M, as, ad, s, d, h, H);
    float ex = __expf(a - fdec(amax[(size_t)d * H + h]));
    float att = ex / (den[(size_t)d * H + h] + 1e-16f);
    const float* xs = xl + (size_t)s * HC + g * 8;
    float* od = out + (size_t)d * HC + g * 8;
#pragma unroll
    for (int i = 0; i < 8; ++i) atomicAdd(&od[i], xs[i] * att);
}

// ---------------- epilogues ----------------
__global__ void elu_bias_k(float* __restrict__ h, const float* __restrict__ b, long total) {
    long t = (long)blockIdx.x * blockDim.x + threadIdx.x;
    if (t >= total) return;
    float v = h[t] + b[t & 63];
    h[t] = v > 0.f ? v : (__expf(v) - 1.f);
}
__global__ void bias_k(float* __restrict__ o, const float* __restrict__ b, long total) {
    long t = (long)blockIdx.x * blockDim.x + threadIdx.x;
    if (t >= total) return;
    o[t] += b[t & 63];
}
__global__ void pool_accum_k(const float* __restrict__ o2, const int* __restrict__ batch,
                             float* __restrict__ pooled, float* __restrict__ gcnt, long total) {
    long t = (long)blockIdx.x * blockDim.x + threadIdx.x;
    if (t >= total) return;
    long n = t >> 6; int c = (int)(t & 63);
    int g = batch[n];
    atomicAdd(&pooled[(size_t)g * HC + c], o2[t]);
    if (c == 0) atomicAdd(&gcnt[g], 1.f);
}
__global__ void pool_div_k(float* __restrict__ pooled, const float* __restrict__ gcnt) {
    int t = blockIdx.x * blockDim.x + threadIdx.x;
    if (t >= GG * HC) return;
    pooled[t] /= fmaxf(gcnt[t >> 6], 1.f);
}

// ---------------- launch ----------------
static inline unsigned gb(long n) { return (unsigned)((n + 255) / 256); }

extern "C" void kernel_launch(void* const* d_in, const int* in_sizes, int n_in,
                              void* d_out, int out_size, void* d_ws, size_t ws_size,
                              hipStream_t stream) {
    (void)in_sizes; (void)n_in; (void)out_size; (void)ws_size;
    const float* x      = (const float*)d_in[0];
    const int*   ei     = (const int*)  d_in[1];
    const int*   batch  = (const int*)  d_in[2];
    const float* eattr  = (const float*)d_in[3];
    const float* W1     = (const float*)d_in[4];
    const float* We1    = (const float*)d_in[5];
    const float* asrc1w = (const float*)d_in[6];
    const float* adst1w = (const float*)d_in[7];
    const float* aedg1w = (const float*)d_in[8];
    const float* b1     = (const float*)d_in[9];
    const float* W2     = (const float*)d_in[10];
    const float* We2    = (const float*)d_in[11];
    const float* asrc2w = (const float*)d_in[12];
    const float* adst2w = (const float*)d_in[13];
    const float* aedg2w = (const float*)d_in[14];
    const float* b2     = (const float*)d_in[15];
    float* pooled = (float*)d_out;

    // workspace layout (~96 MB of f32)
    float* w = (float*)d_ws;
    size_t off = 0;
    float* loop_attr = w + off; off += (size_t)NN * EDIM;   // adjacent to deg (joint zero-fill)
    float* deg       = w + off; off += NN;
    float* xl        = w + off; off += (size_t)NN * HC;     // xl1, then reused as xl2
    float* hbuf      = w + off; off += (size_t)NN * HC;     // conv1 out -> ELU -> layer-2 input
    float* o2        = w + off; off += (size_t)NN * HC;     // conv2 out
    float* as_       = w + off; off += (size_t)NN * 8;
    float* ad_       = w + off; off += (size_t)NN * 8;
    unsigned* amax   = (unsigned*)(w + off); off += (size_t)NN * 8;  // adjacent to den
    float* den       = w + off; off += (size_t)NN * 8;
    float* Mbuf      = w + off; off += 128;
    float* gcnt      = w + off; off += 64;

    const long Etot = (long)EE + NN;
    const unsigned gemm_blocks = (NN + 127) / 128;   // 8 row-tiles (8 waves) per block

    // ---- self loops: mean incoming edge_attr per dst ----
    fill_zero<<<gb((long)NN * (EDIM + 1)), 256, 0, stream>>>(loop_attr, (long)NN * (EDIM + 1));
    sl_accum<<<gb((long)EE * EDIM), 256, 0, stream>>>(ei, eattr, loop_attr, (long)EE * EDIM);
    deg_accum<<<gb(EE), 256, 0, stream>>>(ei, deg, EE);
    sl_div<<<gb((long)NN * EDIM), 256, 0, stream>>>(loop_attr, deg, (long)NN * EDIM);

    // ================= layer 1 (H=8, C=8) =================
    gemm_wmma<FIN><<<gemm_blocks, 256, 0, stream>>>(x, W1, xl, NN);
    node_alpha<<<gb((long)NN * 8), 256, 0, stream>>>(xl, asrc1w, adst1w, as_, ad_, NN, 8, 8);
    make_M<<<1, 128, 0, stream>>>(We1, aedg1w, Mbuf, 8, 8);
    fill_zero<<<gb((long)NN * 16), 256, 0, stream>>>((float*)amax, (long)NN * 16); // amax+den
    fill_zero<<<gb((long)NN * HC), 256, 0, stream>>>(hbuf, (long)NN * HC);
    edge_amax_k<<<gb(Etot * 8), 256, 0, stream>>>(ei, eattr, loop_attr, Mbuf, as_, ad_, amax, Etot, 8);
    edge_den_k <<<gb(Etot * 8), 256, 0, stream>>>(ei, eattr, loop_attr, Mbuf, as_, ad_, amax, den, Etot, 8);
    edge_agg_k <<<gb(Etot * 8), 256, 0, stream>>>(ei, eattr, loop_attr, Mbuf, as_, ad_, amax, den,
                                                  xl, hbuf, Etot, 8);
    elu_bias_k<<<gb((long)NN * HC), 256, 0, stream>>>(hbuf, b1, (long)NN * HC);

    // ================= layer 2 (H=1, C=64) =================
    gemm_wmma<HC><<<gemm_blocks, 256, 0, stream>>>(hbuf, W2, xl, NN);
    node_alpha<<<gb((long)NN), 256, 0, stream>>>(xl, asrc2w, adst2w, as_, ad_, NN, 1, 64);
    make_M<<<1, 128, 0, stream>>>(We2, aedg2w, Mbuf, 1, 64);
    fill_zero<<<gb((long)NN * 2), 256, 0, stream>>>((float*)amax, (long)NN * 2);   // amax(N)+pad
    fill_zero<<<gb((long)NN), 256, 0, stream>>>(den, (long)NN);
    fill_zero<<<gb((long)NN * HC), 256, 0, stream>>>(o2, (long)NN * HC);
    edge_amax_k<<<gb(Etot), 256, 0, stream>>>(ei, eattr, loop_attr, Mbuf, as_, ad_, amax, Etot, 1);
    edge_den_k <<<gb(Etot), 256, 0, stream>>>(ei, eattr, loop_attr, Mbuf, as_, ad_, amax, den, Etot, 1);
    edge_agg_k <<<gb(Etot * 8), 256, 0, stream>>>(ei, eattr, loop_attr, Mbuf, as_, ad_, amax, den,
                                                  xl, o2, Etot, 1);
    bias_k<<<gb((long)NN * HC), 256, 0, stream>>>(o2, b2, (long)NN * HC);

    // ================= mean pool over batch =================
    fill_zero<<<gb(GG * HC), 256, 0, stream>>>(pooled, GG * HC);
    fill_zero<<<1, 64, 0, stream>>>(gcnt, 64);
    pool_accum_k<<<gb((long)NN * HC), 256, 0, stream>>>(o2, batch, pooled, gcnt, (long)NN * HC);
    pool_div_k<<<gb(GG * HC), 256, 0, stream>>>(pooled, gcnt);
}
